// TFJSP_Decoder_hetero_attention_v2_21414706937955
// MI455X (gfx1250) — matrix operand order repaired and checked by reading, not compiled, not run on hardware
//
#include <hip/hip_runtime.h>
#include <hip/hip_bf16.h>
#include <cmath>
#include <cstdint>

typedef _Float16 h16;
typedef __attribute__((ext_vector_type(2)))  __fp16   h2;   // matches cvt_pkrtz return type
typedef __attribute__((ext_vector_type(16))) _Float16 v16h;
typedef __attribute__((ext_vector_type(8)))  float    v8f;

static constexpr float SCALE_QK   = 0.17677669529663687f;   // 1/sqrt(32)
static constexpr float CLIPV      = 10.0f;
static constexpr float INV_SQRT_E = 1.0f / 22.627416997969522f;

// ---------------- WMMA fragment helpers (wave32, f16 16x16x32) ----------------
// frag_a A 16x32 f16, contiguous k (not AMD's table order): lanes 0-15 hold M=0..15 with K 0..15;
// lanes 16-31 hold M=0..15 with K 16..31. VGPR j packs K pairs.
__device__ __forceinline__ v16h frag_a(const h16* s, int m0, int k0, int stride) {
  int l = threadIdx.x & 31;
  const h16* p = s + (m0 + (l & 15)) * stride + k0 + ((l < 16) ? 0 : 16);
  v16h a;
#pragma unroll
  for (int i = 0; i < 8; ++i) a[i] = p[i];
#pragma unroll
  for (int i = 0; i < 8; ++i) a[8 + i] = p[8 + i];
  return a;
}

// frag_b B 32x16 f16, same contiguous k as frag_a: lane n=l&15 holds column n; lanes 0-15 K=0..15,
// lanes 16-31 K=16..31; 16 contiguous halves per lane from [n][k] storage.
__device__ __forceinline__ v16h frag_b(const h16* s, int n0, int k0, int stride) {
  int l = threadIdx.x & 31;
  const h16* p = s + (n0 + (l & 15)) * stride + k0 + ((l < 16) ? 0 : 16);
  v16h b;
#pragma unroll
  for (int i = 0; i < 16; ++i) b[i] = p[i];
  return b;
}

// B-fragment where the upper K half re-reads the lower half; used together with
// an A operand whose K=16..31 columns are zero (so those products vanish).
__device__ __forceinline__ v16h frag_b_dup16(const h16* s, int n0, int stride) {
  int l = threadIdx.x & 31;
  const h16* p = s + (n0 + (l & 15)) * stride;
  v16h b;
#pragma unroll
  for (int i = 0; i < 16; ++i) b[i] = p[i];
  return b;
}

__device__ __forceinline__ v8f wmma16(v16h a, v16h b, v8f c) {
  return __builtin_amdgcn_wmma_f32_16x16x32_f16(false, a, false, b, (short)0, c,
                                                false, false);
}

// ---------------- Kernel: Y[B,512] = X[B,512] @ W[512,512] (+bias) ----------------
__global__ __launch_bounds__(256) void gemm512_kernel(
    const float* __restrict__ X, const float* __restrict__ W,
    const float* __restrict__ bias, float* __restrict__ Y) {
  __shared__ h16 Xs[64 * 40];   // [m][k]
  __shared__ h16 Ws[64 * 40];   // [n][k] (transposed from W[k][n])
  int tid = threadIdx.x;
  int n0 = blockIdx.x * 64, m0 = blockIdx.y * 64;
  int w = tid >> 5, msub = w & 3, nsub0 = (w >> 2) * 2;
  v8f acc0 = {}, acc1 = {};
  for (int kb = 0; kb < 16; ++kb) {
    // X tile 64x32: 512 float4 chunks, packed f16 stores
#pragma unroll
    for (int it = 0; it < 2; ++it) {
      int idx = tid + it * 256;
      int r = idx >> 3, c4 = (idx & 7) * 4;
      float4 f = *(const float4*)(X + (m0 + r) * 512 + kb * 32 + c4);
      *(h2*)&Xs[r * 40 + c4]     = __builtin_amdgcn_cvt_pkrtz(f.x, f.y);
      *(h2*)&Xs[r * 40 + c4 + 2] = __builtin_amdgcn_cvt_pkrtz(f.z, f.w);
    }
    // W tile 32(k)x64(n): coalesced float4 along n, transposed b16 scatter
#pragma unroll
    for (int it = 0; it < 2; ++it) {
      int idx = tid + it * 256;
      int r = idx >> 4, c4 = (idx & 15) * 4;
      float4 f = *(const float4*)(W + (kb * 32 + r) * 512 + n0 + c4);
      Ws[(c4 + 0) * 40 + r] = (h16)f.x;
      Ws[(c4 + 1) * 40 + r] = (h16)f.y;
      Ws[(c4 + 2) * 40 + r] = (h16)f.z;
      Ws[(c4 + 3) * 40 + r] = (h16)f.w;
    }
    __syncthreads();
    v16h a = frag_a(Xs, msub * 16, 0, 40);
    acc0 = wmma16(a, frag_b(Ws, nsub0 * 16, 0, 40), acc0);
    acc1 = wmma16(a, frag_b(Ws, (nsub0 + 1) * 16, 0, 40), acc1);
    __syncthreads();
  }
  int l = tid & 31, rr = (l < 16) ? 0 : 8, cc = l & 15;
#pragma unroll
  for (int r = 0; r < 8; ++r) {
    int row = m0 + msub * 16 + r + rr;
    int c0 = n0 + nsub0 * 16 + cc, c1 = c0 + 16;
    Y[row * 512 + c0] = acc0[r] + (bias ? bias[c0] : 0.f);
    Y[row * 512 + c1] = acc1[r] + (bias ? bias[c1] : 0.f);
  }
}

// -------- Kernel: A[b][h][e] = scale * sum_d q[b][h*32+d] * WK[e][h*32+d] --------
__global__ __launch_bounds__(256) void headA_kernel(
    const float* __restrict__ Xq, const float* __restrict__ WK,
    h16* __restrict__ Aout) {
  __shared__ h16 Xs[64 * 40];   // [b][d]
  __shared__ h16 Ws[64 * 40];   // [e][d]
  int tid = threadIdx.x;
  int e0 = blockIdx.x * 64, b0 = blockIdx.y * 64, h = blockIdx.z;
#pragma unroll
  for (int it = 0; it < 2; ++it) {
    int idx = tid + it * 256;
    int r = idx >> 3, c4 = (idx & 7) * 4;
    float4 fx = *(const float4*)(Xq + (b0 + r) * 512 + h * 32 + c4);
    *(h2*)&Xs[r * 40 + c4] =
        __builtin_amdgcn_cvt_pkrtz(fx.x * SCALE_QK, fx.y * SCALE_QK);
    *(h2*)&Xs[r * 40 + c4 + 2] =
        __builtin_amdgcn_cvt_pkrtz(fx.z * SCALE_QK, fx.w * SCALE_QK);
    float4 fw = *(const float4*)(WK + (e0 + r) * 512 + h * 32 + c4);
    *(h2*)&Ws[r * 40 + c4]     = __builtin_amdgcn_cvt_pkrtz(fw.x, fw.y);
    *(h2*)&Ws[r * 40 + c4 + 2] = __builtin_amdgcn_cvt_pkrtz(fw.z, fw.w);
  }
  __syncthreads();
  int w = tid >> 5, msub = w & 3, nsub0 = (w >> 2) * 2;
  v8f acc0 = {}, acc1 = {};
  v16h a = frag_a(Xs, msub * 16, 0, 40);
  acc0 = wmma16(a, frag_b(Ws, nsub0 * 16, 0, 40), acc0);
  acc1 = wmma16(a, frag_b(Ws, (nsub0 + 1) * 16, 0, 40), acc1);
  int l = tid & 31, rr = (l < 16) ? 0 : 8, cc = l & 15;
#pragma unroll
  for (int r = 0; r < 8; ++r) {
    int b = b0 + msub * 16 + r + rr;
    Aout[(b * 16 + h) * 512 + e0 + nsub0 * 16 + cc]       = (h16)acc0[r];
    Aout[(b * 16 + h) * 512 + e0 + (nsub0 + 1) * 16 + cc] = (h16)acc1[r];
  }
}

// -------- Kernel: out[b][h*32+d] = sum_e U[b][h][e] * WV[e][h*32+d] --------
__global__ __launch_bounds__(256) void headV_kernel(
    const h16* __restrict__ U, const float* __restrict__ WV,
    float* __restrict__ Yout) {
  __shared__ h16 Us[64 * 40];   // [b][e]
  __shared__ h16 Wt[32 * 40];   // [d][e]
  int tid = threadIdx.x;
  int b0 = blockIdx.x * 64, h = blockIdx.y;
  int w = tid >> 5, msub = w & 3, nsub = w >> 2;
  v8f acc = {};
  for (int kb = 0; kb < 16; ++kb) {
#pragma unroll
    for (int it = 0; it < 2; ++it) {        // Us copy: 4 halves per chunk
      int idx = tid + it * 256;
      int r = idx >> 3, c4 = (idx & 7) * 4;
      *(uint2*)&Us[r * 40 + c4] =
          *(const uint2*)(U + ((size_t)(b0 + r) * 16 + h) * 512 + kb * 32 + c4);
    }
    {                                       // Wt transpose: 256 float4 chunks
      int r = tid >> 3, c4 = (tid & 7) * 4; // r: e 0..31, c4: d
      float4 f = *(const float4*)(WV + (kb * 32 + r) * 512 + h * 32 + c4);
      Wt[(c4 + 0) * 40 + r] = (h16)f.x;
      Wt[(c4 + 1) * 40 + r] = (h16)f.y;
      Wt[(c4 + 2) * 40 + r] = (h16)f.z;
      Wt[(c4 + 3) * 40 + r] = (h16)f.w;
    }
    __syncthreads();
    acc = wmma16(frag_a(Us, msub * 16, 0, 40), frag_b(Wt, nsub * 16, 0, 40), acc);
    __syncthreads();
  }
  int l = tid & 31, rr = (l < 16) ? 0 : 8, cc = l & 15;
#pragma unroll
  for (int r = 0; r < 8; ++r) {
    int b = b0 + msub * 16 + r + rr;
    Yout[b * 512 + h * 32 + nsub * 16 + cc] = acc[r];
  }
}

// -------- Kernel: ninf1[b][n] = any_j mask[b][j][n] ? 0 : -inf --------
__global__ __launch_bounds__(64) void maskma_kernel(
    const unsigned char* __restrict__ mask, float* __restrict__ ninf1) {
  int b = blockIdx.x, n = threadIdx.x;
  unsigned any = 0;
  for (int j = 0; j < 128; ++j) any |= mask[b * 8192 + j * 64 + n];
  ninf1[b * 64 + n] = any ? 0.0f : -INFINITY;
}

// -------- Kernel: fused single-query attention core (online softmax) --------
// scores[h,n] = A[b,h,:].enc[b,n,:] + ninf[b,n];  U[b,h,:] = softmax_n(scores).enc
__global__ __launch_bounds__(256) void attncore_kernel(
    const h16* __restrict__ Aall, const float* __restrict__ enc,
    const float* __restrict__ ninf, h16* __restrict__ Uout, int N) {
  __shared__ h16 As[16 * 528];     // A1 [h][e] f16, resident
  __shared__ h16 es[16 * 528];     // enc tile [n][e]   (scores B operand)
  __shared__ h16 esT[512 * 24];    // enc tile [e][n]   (u B operand)
  __shared__ h16 wA[16 * 40];      // softmax weights [h][n], cols 16..39 = 0
  __shared__ float sc[256];        // score tile 16x16
  __shared__ float mrun[16], srun[16], alpha[16];
  int tid = threadIdx.x, b = blockIdx.x;
  int w = tid >> 5, l = tid & 31;
#pragma unroll
  for (int it = 0; it < 4; ++it) {          // A preload: 8-half chunks
    int i = tid + it * 256;
    int r = i >> 6, c8 = (i & 63) * 8;
    *(uint4*)&As[r * 528 + c8] = *(const uint4*)(Aall + (size_t)b * 8192 + i * 8);
  }
  if (tid < 16) {
    mrun[tid] = -INFINITY; srun[tid] = 0.f;
    for (int c = 16; c < 40; ++c) wA[tid * 40 + c] = (h16)0.f;
  }
  v8f acc[4] = {{}, {}, {}, {}};
  int ntiles = N >> 4;
  for (int t = 0; t < ntiles; ++t) {
    __syncthreads();
    sc[tid] = 0.f;
#pragma unroll
    for (int it = 0; it < 8; ++it) {        // 16x512 tile as float4 chunks
      int idx = tid + it * 256;
      int r = idx >> 7, c4 = (idx & 127) * 4;
      float4 f = *(const float4*)(enc + ((size_t)b * N + t * 16 + r) * 512 + c4);
      h2 p0 = __builtin_amdgcn_cvt_pkrtz(f.x, f.y);
      h2 p1 = __builtin_amdgcn_cvt_pkrtz(f.z, f.w);
      *(h2*)&es[r * 528 + c4]     = p0;
      *(h2*)&es[r * 528 + c4 + 2] = p1;
      esT[(c4 + 0) * 24 + r] = (h16)p0[0];
      esT[(c4 + 1) * 24 + r] = (h16)p0[1];
      esT[(c4 + 2) * 24 + r] = (h16)p1[0];
      esT[(c4 + 3) * 24 + r] = (h16)p1[1];
    }
    if (t + 1 < ntiles) {                   // prefetch next 32KB tile
      const char* np = (const char*)(enc + ((size_t)b * N + (t + 1) * 16) * 512);
      __builtin_prefetch(np + tid * 128, 0, 1);
    }
    __syncthreads();
    if (w < 4) {                            // scores: K=512 split over 4 waves
      v8f p = {};
#pragma unroll
      for (int kk = 0; kk < 4; ++kk) {
        int k0 = (w * 4 + kk) * 32;
        p = wmma16(frag_a(As, 0, k0, 528), frag_b(es, 0, k0, 528), p);
      }
      int rr = (l < 16) ? 0 : 8, cc = l & 15;
#pragma unroll
      for (int r = 0; r < 8; ++r) atomicAdd(&sc[(r + rr) * 16 + cc], p[r]);
    }
    __syncthreads();
    if (tid < 16) {                         // online-softmax update per head
      int hh = tid;
      float v[16], mt = -INFINITY;
#pragma unroll
      for (int n = 0; n < 16; ++n) {
        v[n] = sc[hh * 16 + n] + ninf[b * N + t * 16 + n];
        mt = fmaxf(mt, v[n]);
      }
      float mn = fmaxf(mrun[hh], mt);
      if (mn == -INFINITY) {
        for (int n = 0; n < 16; ++n) wA[hh * 40 + n] = (h16)0.f;
        alpha[hh] = 1.f;
      } else {
        float al = expf(mrun[hh] - mn), ssum = 0.f;
#pragma unroll
        for (int n = 0; n < 16; ++n) {
          float e = expf(v[n] - mn);
          wA[hh * 40 + n] = (h16)e; ssum += e;
        }
        srun[hh] = srun[hh] * al + ssum;
        mrun[hh] = mn; alpha[hh] = al;
      }
    }
    __syncthreads();
    {                                       // rescale + accumulate u = w @ enc
      int rr = (l < 16) ? 0 : 8;
      float av[8];
#pragma unroll
      for (int r = 0; r < 8; ++r) av[r] = alpha[r + rr];
      v16h aw = frag_a(wA, 0, 0, 40);
#pragma unroll
      for (int j = 0; j < 4; ++j) {
#pragma unroll
        for (int r = 0; r < 8; ++r) acc[j][r] *= av[r];
        acc[j] = wmma16(aw, frag_b_dup16(esT, (w * 4 + j) * 16, 24), acc[j]);
      }
    }
  }
  __syncthreads();
  int rr = (l < 16) ? 0 : 8, cc = l & 15;
#pragma unroll
  for (int j = 0; j < 4; ++j) {
    int e = (w * 4 + j) * 16 + cc;
#pragma unroll
    for (int r = 0; r < 8; ++r) {
      int hh = r + rr;
      float s = srun[hh];
      Uout[(b * 16 + hh) * 512 + e] = (h16)((s > 0.f) ? acc[j][r] / s : 0.f);
    }
  }
}

// -------- Kernel: score = clip*tanh(mh.enc/sqrtE)+mask; softmax; argmax; gathers --------
__global__ __launch_bounds__(256) void scoresel_kernel(
    const float* __restrict__ mh, const float* __restrict__ enc,
    const float* __restrict__ ninf, const float* __restrict__ embed_nodes,
    int N, int embOff, const unsigned char* __restrict__ mask,
    float* __restrict__ probOut, float* __restrict__ embSelOut,
    float* __restrict__ ninf2Out, const float* __restrict__ addVec,
    float* __restrict__ finalOut) {
  __shared__ float mhs[512];
  __shared__ float et[16 * 516];
  __shared__ float scoreL[128];
  __shared__ float mred, sred;
  __shared__ int selS;
  int tid = threadIdx.x, b = blockIdx.x;
  if (tid < 128)
    *(float4*)&mhs[tid * 4] = *(const float4*)(mh + b * 512 + tid * 4);
  int ntiles = N >> 4;
  for (int t = 0; t < ntiles; ++t) {
    __syncthreads();
#pragma unroll
    for (int it = 0; it < 8; ++it) {
      int idx = tid + it * 256;
      int r = idx >> 7, c4 = (idx & 127) * 4;
      *(float4*)&et[r * 516 + c4] =
          *(const float4*)(enc + ((size_t)b * N + t * 16 + r) * 512 + c4);
    }
    if (t + 1 < ntiles) {
      const char* np = (const char*)(enc + ((size_t)b * N + (t + 1) * 16) * 512);
      __builtin_prefetch(np + tid * 128, 0, 1);
    }
    __syncthreads();
    int nl = tid >> 4, ch = tid & 15;
    float p = 0.f;
#pragma unroll
    for (int i = 0; i < 32; ++i) p += mhs[ch * 32 + i] * et[nl * 516 + ch * 32 + i];
#pragma unroll
    for (int off = 8; off > 0; off >>= 1) p += __shfl_xor(p, off, 32);
    if (ch == 0) scoreL[t * 16 + nl] = p;
  }
  __syncthreads();
  if (tid < N)
    scoreL[tid] = CLIPV * tanhf(scoreL[tid] * INV_SQRT_E) + ninf[b * N + tid];
  __syncthreads();
  if (tid == 0) {
    float m = -INFINITY; int sel = 0;
    for (int n = 0; n < N; ++n) if (scoreL[n] > m) { m = scoreL[n]; sel = n; }
    mred = m; selS = sel;
  }
  __syncthreads();
  float ex = 0.f;
  if (tid < N) {
    float m = mred;
    ex = (m == -INFINITY) ? 0.f : expf(scoreL[tid] - m);
    scoreL[tid] = ex;
  }
  __syncthreads();
  if (tid == 0) {
    float s = 0.f;
    for (int n = 0; n < N; ++n) s += scoreL[n];
    sred = (s > 0.f) ? 1.f / s : 0.f;
  }
  __syncthreads();
  if (tid < N) probOut[b * N + tid] = ex * sred;
  int sel = selS;
  if (embSelOut && tid < 128) {
    const float* src = embed_nodes + ((size_t)b * 192 + embOff + sel) * 512;
    *(float4*)&embSelOut[b * 512 + tid * 4] = *(const float4*)(src + tid * 4);
  }
  if (ninf2Out && tid < 128)
    ninf2Out[b * 128 + tid] = mask[b * 8192 + tid * 64 + sel] ? 0.f : -INFINITY;
  if (finalOut && tid < 128) {
    const float* src = embed_nodes + ((size_t)b * 192 + embOff + sel) * 512;
    float4 a = *(const float4*)(src + tid * 4);
    float4 v = *(const float4*)(addVec + b * 512 + tid * 4);
    a.x += v.x; a.y += v.y; a.z += v.z; a.w += v.w;
    *(float4*)&finalOut[b * 512 + tid * 4] = a;
  }
}

extern "C" void kernel_launch(void* const* d_in, const int* in_sizes, int n_in,
                              void* d_out, int out_size, void* d_ws, size_t ws_size,
                              hipStream_t stream) {
  (void)n_in; (void)out_size; (void)ws_size;
  const float* embed_nodes   = (const float*)d_in[0];
  const float* embed_context = (const float*)d_in[1];
  const float* encoded_opes  = (const float*)d_in[2];
  const float* encoded_mas   = (const float*)d_in[3];
  const unsigned char* mask  = (const unsigned char*)d_in[4];
  const float* Wq2 = (const float*)d_in[5];
  const float* Wq3 = (const float*)d_in[6];
  const float* Wk  = (const float*)d_in[7];
  const float* Wv  = (const float*)d_in[8];
  const float* Wk2 = (const float*)d_in[9];
  const float* Wv2 = (const float*)d_in[10];
  const float* Wc  = (const float*)d_in[11];
  const float* bc  = (const float*)d_in[12];
  const float* Wc2 = (const float*)d_in[13];
  const float* bc2 = (const float*)d_in[14];
  int B = in_sizes[1] / 512;

  char* ws = (char*)d_ws;
  size_t off = 0;
  auto carve = [&](size_t bytes) {
    char* p = ws + off;
    off += (bytes + 255) & ~(size_t)255;
    return p;
  };
  float* q      = (float*)carve((size_t)B * 512 * 4);
  h16*   Af     = (h16*)  carve((size_t)B * 16 * 512 * 2);
  h16*   Uf     = (h16*)  carve((size_t)B * 16 * 512 * 2);
  float* outp   = (float*)carve((size_t)B * 512 * 4);
  float* mhb    = (float*)carve((size_t)B * 512 * 4);
  float* ninf1  = (float*)carve((size_t)B * 64 * 4);
  float* ninf2  = (float*)carve((size_t)B * 128 * 4);
  float* embsel = (float*)carve((size_t)B * 512 * 4);

  float* out_ma  = (float*)d_out;
  float* out_job = out_ma + (size_t)B * 64;
  float* out_emb = out_job + (size_t)B * 128;

  dim3 blk(256);
  dim3 gemmGrid(8, B / 64);
  dim3 headAGrid(8, B / 64, 16);
  dim3 headVGrid(B / 64, 16);

  // ---- stage 1: machine attention ----
  maskma_kernel<<<B, 64, 0, stream>>>(mask, ninf1);
  gemm512_kernel<<<gemmGrid, blk, 0, stream>>>(embed_context, Wq2, nullptr, q);
  headA_kernel<<<headAGrid, blk, 0, stream>>>(q, Wk2, Af);
  attncore_kernel<<<B, blk, 0, stream>>>(Af, encoded_mas, ninf1, Uf, 64);
  headV_kernel<<<headVGrid, blk, 0, stream>>>(Uf, Wv2, outp);
  gemm512_kernel<<<gemmGrid, blk, 0, stream>>>(outp, Wc2, bc2, mhb);
  scoresel_kernel<<<B, blk, 0, stream>>>(mhb, encoded_mas, ninf1, embed_nodes,
                                         64, 128, mask,
                                         out_ma, embsel, ninf2, nullptr, nullptr);
  // ---- stage 2: job attention (q from selected machine embedding) ----
  gemm512_kernel<<<gemmGrid, blk, 0, stream>>>(embsel, Wq3, nullptr, q);
  headA_kernel<<<headAGrid, blk, 0, stream>>>(q, Wk, Af);
  attncore_kernel<<<B, blk, 0, stream>>>(Af, encoded_opes, ninf2, Uf, 128);
  headV_kernel<<<headVGrid, blk, 0, stream>>>(Uf, Wv, outp);
  gemm512_kernel<<<gemmGrid, blk, 0, stream>>>(outp, Wc, bc, mhb);
  scoresel_kernel<<<B, blk, 0, stream>>>(mhb, encoded_opes, ninf2, embed_nodes,
                                         128, 0, nullptr,
                                         out_job, nullptr, nullptr, embsel, out_emb);
}
